// GnnRnnAgent_42210938585345
// MI455X (gfx1250) — matrix operand order repaired and checked by reading, not compiled, not run on hardware
//
#include <hip/hip_runtime.h>
#include <hip/hip_bf16.h>

// ---------------- problem constants (from reference) ----------------
static constexpr int NN = 100000;   // nodes
static constexpr int EE = 1600000;  // edges
static constexpr int IN = 128;
static constexpr int HH = 256;
static constexpr int GG = 128;
static constexpr int MM = 128;
static constexpr int AA = 16;

typedef __attribute__((ext_vector_type(16))) __bf16 bf16x16;
typedef __attribute__((ext_vector_type(8)))  float  floatx8;

// ---------------- bf16 helpers (bit-exact, no libcall risk) ----------------
__device__ __forceinline__ __bf16 f2bf(float f) {
    unsigned u = __float_as_uint(f);
    unsigned r = (u + 0x7FFFu + ((u >> 16) & 1u)) >> 16;   // round-to-nearest-even
    unsigned short s = (unsigned short)r;
    return __builtin_bit_cast(__bf16, s);
}
__device__ __forceinline__ float bf2f(__bf16 b) {
    unsigned short s = __builtin_bit_cast(unsigned short, b);
    return __uint_as_float(((unsigned)s) << 16);
}

// monotonic float<->uint mapping for atomic segment-max
__device__ __forceinline__ unsigned fkey(float f) {
    unsigned u = __float_as_uint(f);
    return (u & 0x80000000u) ? ~u : (u | 0x80000000u);
}
__device__ __forceinline__ float funkey(unsigned u) {
    return __uint_as_float((u & 0x80000000u) ? (u ^ 0x80000000u) : ~u);
}

// =====================================================================
// WMMA GEMM:  C[nrows,ncols] = act( A[nrows,K] * B[K,ncols] + bias )
// 128x64 block tile, BK=32, 128 threads = 4 waves.
// Each wave: 32 rows x 64 cols = 2x4 WMMA tiles -> B frags reused across
// two A frags (24 wmma per chunk vs 192 frag-element LDS reads, ~4:1
// after b32 pairing), keeping the XDL pipe fed.
// bf16 hi/lo split (3 wmma per 16x16x32 step) for ~fp32 accuracy.
// =====================================================================
__global__ __launch_bounds__(128)
void wmma_gemm(const float* __restrict__ A, const float* __restrict__ B,
               const float* __restrict__ bias, float* __restrict__ C,
               int nrows, int K, int ncols, int dorelu)
{
    __shared__ __bf16 Ah[128][34];  // padded: 17-bank stride, conflict free
    __shared__ __bf16 Al[128][34];
    __shared__ __bf16 Bh[64][34];   // transposed: [col][k]
    __shared__ __bf16 Bl[64][34];

    const int tid  = threadIdx.x;
    const int wave = tid >> 5;
    const int lane = tid & 31;
    const int lr   = lane & 15;
    const int hi   = lane >> 4;
    const int row0 = blockIdx.x * 128;
    const int col0 = blockIdx.y * 64;

    floatx8 acc[2][4];
#pragma unroll
    for (int g = 0; g < 2; ++g)
#pragma unroll
        for (int t = 0; t < 4; ++t)
#pragma unroll
            for (int e = 0; e < 8; ++e) acc[g][t][e] = 0.0f;

    for (int kc = 0; kc < K; kc += 32) {
        // ---- stage A tile (128x32 f32, coalesced: 32 elems/thread) ----
        for (int it = 0; it < 32; ++it) {
            int i = tid + it * 128;
            int r = i >> 5, k = i & 31;
            int gr = row0 + r;
            float v = 0.0f;
            if (gr < nrows) {
                v = A[(long)gr * K + kc + k];
                if (kc + 32 < K)
                    __builtin_prefetch(&A[(long)gr * K + kc + 32 + k], 0, 3);
            }
            __bf16 h = f2bf(v);
            Ah[r][k] = h;
            Al[r][k] = f2bf(v - bf2f(h));
        }
        // ---- stage B tile (32x64 f32 -> transposed LDS) ----
#pragma unroll
        for (int it = 0; it < 16; ++it) {
            int i = tid + it * 128;
            int k = i >> 6, c = i & 63;
            int gc = col0 + c;
            float v = (gc < ncols) ? B[(long)(kc + k) * ncols + gc] : 0.0f;
            __bf16 h = f2bf(v);
            Bh[c][k] = h;
            Bl[c][k] = f2bf(v - bf2f(h));
        }
        __syncthreads();

        // ---- two A fragments (ISA: 16-bit A 16x32 layout) ----
        bf16x16 a_h[2], a_l[2];
#pragma unroll
        for (int g = 0; g < 2; ++g)
#pragma unroll
            for (int e = 0; e < 16; ++e) {
                int kk = e + ((e >= 8) ? 8 : 0) + hi * 8;
                a_h[g][e] = Ah[wave * 32 + g * 16 + lr][kk];
                a_l[g][e] = Al[wave * 32 + g * 16 + lr][kk];
            }
        // ---- 4 N-tiles x 2 M-tiles, 3 wmma each (hh + hl + lh) ----
#pragma unroll
        for (int nt = 0; nt < 4; ++nt) {
            bf16x16 b_h, b_l;
#pragma unroll
            for (int e = 0; e < 16; ++e) {
                int kk = e + ((e >= 8) ? 8 : 0) + hi * 8;
                b_h[e] = Bh[nt * 16 + lr][kk];
                b_l[e] = Bl[nt * 16 + lr][kk];
            }
#pragma unroll
            for (int g = 0; g < 2; ++g) {
                acc[g][nt] = __builtin_amdgcn_wmma_f32_16x16x32_bf16(
                    false, a_h[g], false, b_h, (short)0, acc[g][nt], false, false);
                acc[g][nt] = __builtin_amdgcn_wmma_f32_16x16x32_bf16(
                    false, a_h[g], false, b_l, (short)0, acc[g][nt], false, false);
                acc[g][nt] = __builtin_amdgcn_wmma_f32_16x16x32_bf16(
                    false, a_l[g], false, b_h, (short)0, acc[g][nt], false, false);
            }
        }
        __syncthreads();
    }

    // ---- epilogue: C/D layout -> global, fused bias + relu ----
#pragma unroll
    for (int nt = 0; nt < 4; ++nt) {
        int gc = col0 + nt * 16 + lr;
        if (gc >= ncols) continue;
        float bv = bias ? bias[gc] : 0.0f;
#pragma unroll
        for (int g = 0; g < 2; ++g)
#pragma unroll
            for (int r = 0; r < 8; ++r) {
                int gr = row0 + wave * 32 + g * 16 + r + hi * 8;
                if (gr < nrows) {
                    float v = acc[g][nt][r] + bv;
                    if (dorelu) v = fmaxf(v, 0.0f);
                    C[(long)gr * ncols + gc] = v;
                }
            }
    }
}

// =====================================================================
// Edge-phase kernels
// =====================================================================
__global__ void k_deg_attr(const int* __restrict__ ei, const float* __restrict__ ea,
                           float* __restrict__ deg, float* __restrict__ asum)
{
    int e = blockIdx.x * 256 + threadIdx.x;
    if (e >= EE) return;
    int d = ei[EE + e];
    atomicAdd(&deg[d], 1.0f);
    atomicAdd(&asum[d * 3 + 0], ea[e * 3 + 0]);
    atomicAdd(&asum[d * 3 + 1], ea[e * 3 + 1]);
    atomicAdd(&asum[d * 3 + 2], ea[e * 3 + 2]);
}

__global__ void k_attr_mean(float* __restrict__ asum, const float* __restrict__ deg)
{
    int i = blockIdx.x * 256 + threadIdx.x;
    if (i >= NN) return;
    float s = 1.0f / fmaxf(deg[i], 1.0f);
    asum[i * 3 + 0] *= s;
    asum[i * 3 + 1] *= s;
    asum[i * 3 + 2] *= s;
}

// one wave32 per edge: attention logit + segment max
__global__ __launch_bounds__(256)
void k_logits(const int* __restrict__ ei, const float* __restrict__ ea,
              const float* __restrict__ amean, const float* __restrict__ xl,
              const float* __restrict__ xr, const float* __restrict__ We,
              const float* __restrict__ att, float* __restrict__ elog,
              unsigned* __restrict__ emax)
{
    int widx = blockIdx.x * 8 + (threadIdx.x >> 5);
    int lane = threadIdx.x & 31;
    if (widx >= EE + NN) return;
    int s, d; float e0, e1, e2;
    if (widx < EE) {
        s = ei[widx]; d = ei[EE + widx];
        e0 = ea[widx * 3]; e1 = ea[widx * 3 + 1]; e2 = ea[widx * 3 + 2];
    } else {
        s = d = widx - EE;
        e0 = amean[s * 3]; e1 = amean[s * 3 + 1]; e2 = amean[s * 3 + 2];
    }
    const float* pl = xl + (long)s * GG;
    const float* pr = xr + (long)d * GG;
    float acc = 0.0f;
#pragma unroll
    for (int j = 0; j < 4; ++j) {
        int c = lane + 32 * j;
        float m = pl[c] + pr[c] + e0 * We[c] + e1 * We[GG + c] + e2 * We[2 * GG + c];
        m = (m > 0.0f) ? m : 0.2f * m;        // leaky_relu(0.2)
        acc += m * att[c];
    }
#pragma unroll
    for (int off = 16; off; off >>= 1) acc += __shfl_xor(acc, off, 32);
    if (lane == 0) {
        elog[widx] = acc;
        atomicMax(&emax[d], fkey(acc));
    }
}

__global__ void k_exp(const int* __restrict__ ei, float* __restrict__ elog,
                      const unsigned* __restrict__ emax, float* __restrict__ denom)
{
    int i = blockIdx.x * 256 + threadIdx.x;
    if (i >= EE + NN) return;
    int d = (i < EE) ? ei[EE + i] : (i - EE);
    float p = __expf(elog[i] - funkey(emax[d]));
    elog[i] = p;                               // in-place: elog becomes p
    atomicAdd(&denom[d], p);
}

// one wave32 per edge: agg[dst] += alpha * xl[src]
__global__ __launch_bounds__(256)
void k_scatter(const int* __restrict__ ei, const float* __restrict__ p,
               const float* __restrict__ denom, const float* __restrict__ xl,
               float* __restrict__ agg)
{
    int widx = blockIdx.x * 8 + (threadIdx.x >> 5);
    int lane = threadIdx.x & 31;
    if (widx >= EE + NN) return;
    int s, d;
    if (widx < EE) { s = ei[widx]; d = ei[EE + widx]; }
    else           { s = d = widx - EE; }
    float alpha = p[widx] / denom[d];
    const float* pl = xl + (long)s * GG;
    float* pa = agg + (long)d * GG;
#pragma unroll
    for (int j = 0; j < 4; ++j) {
        int c = lane + 32 * j;
        atomicAdd(&pa[c], alpha * pl[c]);
    }
}

__global__ void k_hgat(float* __restrict__ agg, const float* __restrict__ xres,
                       const float* __restrict__ gbias)
{
    int i = blockIdx.x * 256 + threadIdx.x;
    if (i >= NN * GG) return;
    agg[i] = fmaxf(agg[i] + xres[i] + gbias[i & (GG - 1)], 0.0f);
}

__global__ void k_gru(const float* __restrict__ gi, const float* __restrict__ gh,
                      const float* __restrict__ hprev, float* __restrict__ hout)
{
    int i = blockIdx.x * 256 + threadIdx.x;
    if (i >= NN * MM) return;
    int row = i >> 7, c = i & (MM - 1);
    const float* pi = gi + (long)row * (3 * MM);
    const float* ph = gh + (long)row * (3 * MM);
    float r = 1.0f / (1.0f + __expf(-(pi[c] + ph[c])));
    float z = 1.0f / (1.0f + __expf(-(pi[MM + c] + ph[MM + c])));
    float n = tanhf(pi[2 * MM + c] + r * ph[2 * MM + c]);
    hout[i] = (1.0f - z) * n + z * hprev[i];
}

// =====================================================================
// launch
// =====================================================================
static inline void gemm(const float* A, const float* B, const float* bias, float* C,
                        int nrows, int K, int ncols, int relu, hipStream_t s)
{
    dim3 grid((nrows + 127) / 128, (ncols + 63) / 64);
    wmma_gemm<<<grid, 128, 0, s>>>(A, B, bias, C, nrows, K, ncols, relu);
}

extern "C" void kernel_launch(void* const* d_in, const int* in_sizes, int n_in,
                              void* d_out, int out_size, void* d_ws, size_t ws_size,
                              hipStream_t stream)
{
    (void)in_sizes; (void)n_in; (void)out_size; (void)ws_size;
    const float* inputs = (const float*)d_in[0];
    const float* hidden = (const float*)d_in[1];
    const int*   ei     = (const int*)  d_in[2];
    const float* ea     = (const float*)d_in[3];
    const float* W0  = (const float*)d_in[4];  const float* b0  = (const float*)d_in[5];
    const float* W1  = (const float*)d_in[6];  const float* b1  = (const float*)d_in[7];
    const float* Wl  = (const float*)d_in[8];  const float* Wr  = (const float*)d_in[9];
    const float* We  = (const float*)d_in[10]; const float* att = (const float*)d_in[11];
    const float* gbias = (const float*)d_in[12];
    const float* Wres  = (const float*)d_in[13];
    const float* Wih = (const float*)d_in[14]; const float* bih = (const float*)d_in[15];
    const float* Whh = (const float*)d_in[16]; const float* bhh = (const float*)d_in[17];
    const float* Wq  = (const float*)d_in[18]; const float* bq  = (const float*)d_in[19];

    // ---- workspace layout (with deliberate aliasing; ~420 MB) ----
    float* ws   = (float*)d_ws;
    float* x1   = ws;                          // [N,256]  } gh aliases x1+xl = [N,384]
    float* xl   = x1   + (size_t)NN * HH;      // [N,128]  }
    float* x    = xl   + (size_t)NN * GG;      // [N,256]  } gi aliases x+xr  = [N,384]
    float* xr   = x    + (size_t)NN * HH;      // [N,128]  }
    float* xres = xr   + (size_t)NN * GG;      // [N,128]
    float* agg  = xres + (size_t)NN * GG;      // [N,128]  (becomes h_gat in place)
    float* deg  = agg  + (size_t)NN * GG;      // [N]
    float* asum = deg  + NN;                   // [N,3]    (becomes attr_mean in place)
    unsigned* emax = (unsigned*)(asum + 3 * (size_t)NN);  // [N]
    float* denom = (float*)emax + NN;          // [N]
    float* elog  = denom + NN;                 // [E+N]    (becomes p in place)
    float* gh = x1;                            // alias, live after scatter
    float* gi = x;                             // alias, live after scatter

    float* q_out = (float*)d_out;              // [N,16]
    float* h_out = q_out + (size_t)NN * AA;    // [N,128]

    // ---- zero accumulators (memset nodes are graph-capturable) ----
    hipMemsetAsync(deg,   0, sizeof(float) * NN,        stream);
    hipMemsetAsync(asum,  0, sizeof(float) * NN * 3,    stream);
    hipMemsetAsync(emax,  0, sizeof(unsigned) * NN,     stream);
    hipMemsetAsync(denom, 0, sizeof(float) * NN,        stream);
    hipMemsetAsync(agg,   0, sizeof(float) * NN * GG,   stream);

    // ---- MLP ----
    gemm(inputs, W0, b0, x1, NN, IN, HH, 1, stream);          // x1 = relu(in@W0+b0)
    gemm(x1,     W1, b1, x,  NN, HH, HH, 1, stream);          // x  = relu(x1@W1+b1)

    // ---- GAT projections ----
    gemm(x, Wl,   nullptr, xl,   NN, HH, GG, 0, stream);
    gemm(x, Wr,   nullptr, xr,   NN, HH, GG, 0, stream);
    gemm(x, Wres, nullptr, xres, NN, HH, GG, 0, stream);

    // ---- edge phase ----
    k_deg_attr <<<(EE + 255) / 256, 256, 0, stream>>>(ei, ea, deg, asum);
    k_attr_mean<<<(NN + 255) / 256, 256, 0, stream>>>(asum, deg);
    {
        int waves = EE + NN;
        int blk = (waves + 7) / 8;
        k_logits <<<blk, 256, 0, stream>>>(ei, ea, asum, xl, xr, We, att, elog, emax);
        k_exp    <<<(waves + 255) / 256, 256, 0, stream>>>(ei, elog, emax, denom);
        k_scatter<<<blk, 256, 0, stream>>>(ei, elog, denom, xl, agg);
    }
    k_hgat<<<(NN * GG + 255) / 256, 256, 0, stream>>>(agg, xres, gbias);

    // ---- GRU ----
    gemm(agg,    Wih, bih, gi, NN, GG, 3 * MM, 0, stream);
    gemm(hidden, Whh, bhh, gh, NN, MM, 3 * MM, 0, stream);
    k_gru<<<(NN * MM + 255) / 256, 256, 0, stream>>>(gi, gh, hidden, h_out);

    // ---- q = h @ Wq + bq ----
    gemm(h_out, Wq, bq, q_out, NN, MM, AA, 0, stream);
}
